// SDGNN_SGC_28896539968170
// MI455X (gfx1250) — matrix operand order repaired and checked by reading, not compiled, not run on hardware
//
#include <hip/hip_runtime.h>
#include <hip/hip_bf16.h>

#define D_FEAT 256
#define GEMM_NT 4   // 16x64 output strip per wave: A-fragment reused across 4 WMMAs

typedef __attribute__((ext_vector_type(16))) __bf16 v16bf;
typedef __attribute__((ext_vector_type(8)))  __bf16 v8bf;
typedef __attribute__((ext_vector_type(8)))  float  v8f;

#define SHUF16(lo, hi) __builtin_shufflevector(lo, hi, \
    0, 1, 2, 3, 4, 5, 6, 7, 8, 9, 10, 11, 12, 13, 14, 15)

// ---------------- init / scatter ----------------

__global__ void k_zero_f32(float* __restrict__ p, long n) {
    long i = (long)blockIdx.x * blockDim.x + threadIdx.x;
    long s = (long)gridDim.x * blockDim.x;
    for (; i < n; i += s) p[i] = 0.0f;
}

__global__ void k_fill_i32(int* __restrict__ p, int v, int n) {
    int i = blockIdx.x * blockDim.x + threadIdx.x;
    if (i < n) p[i] = v;
}

__global__ void k_scatter_edges(const int* __restrict__ row, const int* __restrict__ col,
                                float* __restrict__ A, float* __restrict__ deg,
                                int E2, int N) {
    int e = blockIdx.x * blockDim.x + threadIdx.x;
    if (e < E2) {
        int r = row[e], c = col[e];
        atomicAdd(&A[(long)r * N + c], 1.0f);
        atomicAdd(&deg[r], 1.0f);
    }
}

__global__ void k_self_loops(float* __restrict__ A, float* __restrict__ deg, int N) {
    int i = blockIdx.x * blockDim.x + threadIdx.x;
    if (i < N) {
        atomicAdd(&A[(long)i * N + i], 1.0f);
        atomicAdd(&deg[i], 1.0f);
    }
}

__global__ void k_deg_to_dis(float* __restrict__ deg, int N) {
    int i = blockIdx.x * blockDim.x + threadIdx.x;
    if (i < N) {
        float d = deg[i];
        deg[i] = (d > 0.0f) ? rsqrtf(d) : 0.0f;   // becomes dis[]
    }
}

// Sb[i][k] = bf16( dis[i] * A[i][k] * dis[k] )  -- row-major bf16 normalized adjacency
__global__ void k_build_S(const float* __restrict__ A, const float* __restrict__ dis,
                          __bf16* __restrict__ Sb, int N) {
    long i = (long)blockIdx.x * blockDim.x + threadIdx.x;
    long s = (long)gridDim.x * blockDim.x;
    const long n2 = (long)N * N;
    for (; i < n2; i += s) {
        int r = (int)(i >> 12);      // N == 4096
        int k = (int)(i & (N - 1));
        Sb[i] = (__bf16)(dis[r] * A[i] * dis[k]);
    }
}

// f32 [K x Dcols] row-major  ->  bf16 WMMA-B-fragment-swizzled:
// dst[ ((kt*nct + ct)*32 + half*16 + colInTile)*16 + ((k&7) + 8*((k>>4)&1)) ], half = (k>>3)&1
// so each lane's 16 K-values for one 16x16 B tile are contiguous (2x b128 loads),
// the 4 tiles of a strip are 1KB apart, and k-steps stride a constant 16KB.
__global__ void k_swizzle_B(const float* __restrict__ src, __bf16* __restrict__ dst,
                            long n, int Dcols) {
    long i = (long)blockIdx.x * blockDim.x + threadIdx.x;
    long s = (long)gridDim.x * blockDim.x;
    const int nct = Dcols >> 4;
    for (; i < n; i += s) {
        int k   = (int)(i / Dcols);
        int col = (int)(i % Dcols);
        int kt   = k >> 5, kin = k & 31;
        int half = (kin >> 3) & 1, e = (kin & 7) | ((kin >> 4) << 3);
        int ct   = col >> 4, cin = col & 15;
        long tile = (long)kt * nct + ct;
        dst[(tile * 32 + half * 16 + cin) * 16 + e] = (__bf16)src[i];
    }
}

// ---------------- WMMA GEMM: Cout[N x Dcols] = Sb @ B ----------------
// 2x-unrolled software pipeline: the two fragment buffers alternate roles
// (load f1 / compute f0 / load f0 / compute f1) so no register-copy rotation
// is needed and every WMMA group runs with the other buffer's loads in flight.

struct Frags {
    v16bf a;
    v16bf b[GEMM_NT];
};

__device__ __forceinline__ void load_frags(const __bf16* ap, const __bf16* bp, Frags& f) {
    f.a = SHUF16(*(const v8bf*)(ap), *(const v8bf*)(ap + 16));
#pragma unroll
    for (int t = 0; t < GEMM_NT; ++t)
        f.b[t] = SHUF16(*(const v8bf*)(bp + t * 512),
                        *(const v8bf*)(bp + t * 512 + 8));
}

__device__ __forceinline__ void do_wmma(const Frags& f, v8f acc[GEMM_NT]) {
#pragma unroll
    for (int t = 0; t < GEMM_NT; ++t)
        acc[t] = __builtin_amdgcn_wmma_f32_16x16x32_bf16(
            false, f.a, false, f.b[t], (short)0, acc[t], false, false);
}

__global__ __launch_bounds__(128) void k_wmma_gemm(
    const __bf16* __restrict__ Sb, const __bf16* __restrict__ Bsw,
    float* __restrict__ Cout, int Kdim, int Dcols)
{
    const int wave  = threadIdx.x >> 5;
    const int lane  = threadIdx.x & 31;
    const int tile  = blockIdx.x * (blockDim.x >> 5) + wave;
    const int ngrps = Dcols / (16 * GEMM_NT);
    const int m0    = (tile / ngrps) * 16;
    const int ng    = (tile % ngrps);
    const int half  = lane >> 4;
    const int lidx  = lane & 15;
    const int nct   = Dcols >> 4;

    const __bf16* ap = Sb + (long)(m0 + lidx) * Kdim + half * 8;      // A frag base
    const __bf16* bp = Bsw + ((long)ng * GEMM_NT * 32 + lane) * 16;   // B strip base
    const long bstep = (long)nct * 32 * 16;   // elements per 32-wide k-step (16KB)

    v8f acc[GEMM_NT];
    const v8f vzero = {0.f, 0.f, 0.f, 0.f, 0.f, 0.f, 0.f, 0.f};
#pragma unroll
    for (int t = 0; t < GEMM_NT; ++t) acc[t] = vzero;

    Frags f0, f1;
    load_frags(ap, bp, f0);                       // k-step 0

    int k0;
    for (k0 = 0; k0 + 64 < Kdim; k0 += 64) {      // steady state, no buffer copies
        __builtin_prefetch(ap + 2048, 0, 1);      // global_prefetch_b8 ahead on S row
        load_frags(ap + 32, bp + bstep, f1);      // k-step k0+32 in flight
        do_wmma(f0, acc);
        load_frags(ap + 64, bp + 2 * bstep, f0);  // k-step k0+64 in flight
        do_wmma(f1, acc);
        ap += 64;
        bp += 2 * bstep;
    }
    // tail: last 64 of K (f0 already loaded)
    load_frags(ap + 32, bp + bstep, f1);
    do_wmma(f0, acc);
    do_wmma(f1, acc);

#pragma unroll
    for (int t = 0; t < GEMM_NT; ++t) {
        const int colN = (ng * GEMM_NT + t) * 16 + lidx;
#pragma unroll
        for (int r = 0; r < 8; ++r)
            Cout[(long)(m0 + r + 8 * half) * Dcols + colN] = acc[t][r];
    }
}

// ---------------- community epilogue ----------------

__global__ void k_counts(const int* __restrict__ comm_ids, int* __restrict__ counts, int M) {
    int i = blockIdx.x * blockDim.x + threadIdx.x;
    if (i < M) atomicAdd(&counts[comm_ids[i]], 1);
}

__global__ void k_sums(const int* __restrict__ member_nodes, const int* __restrict__ comm_ids,
                       const float* __restrict__ prop, float* __restrict__ sums) {
    const int m = blockIdx.x;          // one member per block
    const int d = threadIdx.x;         // D_FEAT threads
    const int node = member_nodes[m];
    const int cid  = comm_ids[m];
    atomicAdd(&sums[(long)cid * D_FEAT + d], prop[(long)node * D_FEAT + d]);
}

__global__ void k_ranges(const int* __restrict__ owner,
                         int* __restrict__ startc, int* __restrict__ endc, int C) {
    int c = blockIdx.x * blockDim.x + threadIdx.x;
    if (c < C) {
        int o = owner[c];
        atomicMin(&startc[o], c);
        atomicMax(&endc[o], c);
    }
}

__global__ void k_final(const float* __restrict__ prop, const float* __restrict__ sums,
                        const int* __restrict__ counts, const int* __restrict__ owner,
                        const int* __restrict__ startc, const int* __restrict__ endc,
                        float* __restrict__ out) {
    const int n = blockIdx.x;
    const int d = threadIdx.x;
    const float p = prop[(long)n * D_FEAT + d];
    out[(long)n * 2 * D_FEAT + d] = fmaxf(p, 0.0f);   // cat[:, :D] == fallback[:, :D] == prop
    const int s = startc[n], e = endc[n];
    float v;
    if (s > e) {                 // no community owned by n -> fallback second half
        v = p;
    } else {                     // segment_max over means of n's (contiguous) communities
        float m = -3.402823466e38f;
        for (int c = s; c <= e; ++c)
            if (owner[c] == n)
                m = fmaxf(m, sums[(long)c * D_FEAT + d] / (float)counts[c]);
        v = m;
    }
    out[(long)n * 2 * D_FEAT + D_FEAT + d] = fmaxf(v, 0.0f);
}

// ---------------- launch ----------------

extern "C" void kernel_launch(void* const* d_in, const int* in_sizes, int n_in,
                              void* d_out, int out_size, void* d_ws, size_t ws_size,
                              hipStream_t stream) {
    const int*   edge         = (const int*)d_in[0];
    const float* X            = (const float*)d_in[1];
    const int*   member_nodes = (const int*)d_in[2];
    const int*   comm_ids     = (const int*)d_in[3];
    const int*   owner        = (const int*)d_in[4];
    float*       out          = (float*)d_out;

    const int E2 = in_sizes[0] / 2;          // directed edge count (both directions present)
    const int N  = in_sizes[1] / D_FEAT;     // 4096
    const int M  = in_sizes[2];              // total community memberships
    const int C  = in_sizes[4];              // number of communities
    const int*   erow = edge;
    const int*   ecol = edge + E2;

    // workspace layout (256B aligned); A(67MB)+Sb(33.5MB) both L2-resident (192MB)
    char* w = (char*)d_ws;
    auto alloc = [&](size_t bytes) { char* p = w; w += (bytes + 255) & ~(size_t)255; return p; };
    float*  A      = (float*)alloc((size_t)N * N * sizeof(float));
    __bf16* Sb     = (__bf16*)alloc((size_t)N * N * sizeof(__bf16));
    float*  dis    = (float*)alloc((size_t)N * sizeof(float));          // deg, then rsqrt in place
    __bf16* Xb     = (__bf16*)alloc((size_t)N * D_FEAT * sizeof(__bf16));
    float*  T      = (float*)alloc((size_t)N * D_FEAT * sizeof(float));
    __bf16* Tb     = (__bf16*)alloc((size_t)N * D_FEAT * sizeof(__bf16));
    float*  prop   = (float*)alloc((size_t)N * D_FEAT * sizeof(float));
    float*  sums   = (float*)alloc((size_t)C * D_FEAT * sizeof(float));
    int*    counts = (int*)alloc((size_t)C * sizeof(int));
    int*    startc = (int*)alloc((size_t)N * sizeof(int));
    int*    endc   = (int*)alloc((size_t)N * sizeof(int));
    (void)ws_size; (void)n_in; (void)out_size;

    // 1) init
    k_zero_f32<<<8192, 256, 0, stream>>>(A, (long)N * N);
    k_zero_f32<<<64, 256, 0, stream>>>(dis, N);
    k_zero_f32<<<4096, 256, 0, stream>>>(sums, (long)C * D_FEAT);
    k_fill_i32<<<(C + 255) / 256, 256, 0, stream>>>(counts, 0, C);
    k_fill_i32<<<(N + 255) / 256, 256, 0, stream>>>(startc, 0x7fffffff, N);
    k_fill_i32<<<(N + 255) / 256, 256, 0, stream>>>(endc, -1, N);

    // 2) build A (+I), degrees, dis = rsqrt(deg), then bf16 normalized adjacency Sb
    k_scatter_edges<<<(E2 + 255) / 256, 256, 0, stream>>>(erow, ecol, A, dis, E2, N);
    k_self_loops<<<(N + 255) / 256, 256, 0, stream>>>(A, dis, N);
    k_deg_to_dis<<<(N + 255) / 256, 256, 0, stream>>>(dis, N);
    k_build_S<<<8192, 256, 0, stream>>>(A, dis, Sb, N);

    // 3) prop = S @ (S @ X); bf16 WMMA, f32 accumulate; B pre-swizzled to fragment order
    k_swizzle_B<<<4096, 256, 0, stream>>>(X, Xb, (long)N * D_FEAT, D_FEAT);
    const int tiles  = (N / 16) * (D_FEAT / (16 * GEMM_NT));   // exact: no EXEC holes
    const int blocks = tiles / 4;                              // 4 waves / 128-thread block
    k_wmma_gemm<<<blocks, 128, 0, stream>>>(Sb, Xb, T, N, D_FEAT);
    k_swizzle_B<<<4096, 256, 0, stream>>>(T, Tb, (long)N * D_FEAT, D_FEAT);
    k_wmma_gemm<<<blocks, 128, 0, stream>>>(Sb, Tb, prop, N, D_FEAT);

    // 4) community segment-mean / segment-max epilogue
    k_counts<<<(M + 255) / 256, 256, 0, stream>>>(comm_ids, counts, M);
    k_sums<<<M, D_FEAT, 0, stream>>>(member_nodes, comm_ids, prop, sums);
    k_ranges<<<(C + 255) / 256, 256, 0, stream>>>(owner, startc, endc, C);
    k_final<<<N, D_FEAT, 0, stream>>>(prop, sums, counts, owner, startc, endc, out);
}
